// ShiftedWindowAttention3D_52243982188741
// MI455X (gfx1250) — matrix operand order repaired and checked by reading, not compiled, not run on hardware
//
#include <hip/hip_runtime.h>

// ---------------------------------------------------------------------------
// SwinV2 3D shifted-window attention, fused per-window kernel for gfx1250.
// One workgroup (256 threads = 8 wave32) per window.  All matmuls use
// v_wmma_f32_16x16x32_f16.  CPB relative-position bias and f16 copies of the
// projection weights are precomputed into d_ws by small one-shot kernels.
// ---------------------------------------------------------------------------

typedef _Float16 v16h __attribute__((ext_vector_type(16)));
typedef float    v8f  __attribute__((ext_vector_type(8)));

#define DD   16
#define HH   112
#define WW_  112
#define CC   128
#define NH   4
#define HDIM 32
#define WD   2
#define WH   7
#define WWIN 7
#define SD   1
#define SH   3
#define SW   3
#define NTOK 98     // 2*7*7
#define NPAD 112    // 7 M-tiles of 16
#define NWIN 2048   // 8*16*16
#define MTAB 507    // 3*13*13

__device__ __forceinline__ v8f vzero8() {
  v8f z;
#pragma unroll
  for (int i = 0; i < 8; ++i) z[i] = 0.0f;
  return z;
}
__device__ __forceinline__ v16h vzero16() {
  v16h z;
#pragma unroll
  for (int i = 0; i < 16; ++i) z[i] = (_Float16)0.0f;
  return z;
}

__device__ __forceinline__ v8f wmma16(v16h a, v16h b, v8f c) {
  return __builtin_amdgcn_wmma_f32_16x16x32_f16(false, a, false, b,
                                                (short)0, c, false, false);
}

// A fragment (16x32 f16, row-major LDS, ldh halves/row).
// lane&15 = row; lane>=16 -> K offset +16.  VGPR v0..7 hold K {0..15}(+off):
// two 16B chunks at p and p+8.
__device__ __forceinline__ v16h ld_a16(const _Float16* base, int row0,
                                       int ldh, int k0) {
  const int lane = threadIdx.x & 31;
  const _Float16* p = base + (row0 + (lane & 15)) * ldh + k0 + ((lane >> 4) << 4);
  union { v16h v; uint4 q[2]; } u;
  u.q[0] = *(const uint4*)(p);
  u.q[1] = *(const uint4*)(p + 8);
  return u.v;
}

// B fragment (32x16 f16) where B[kk][n] = base[(ncol0+n)*ldh + k0 + kk].
// lane&15 = column n; lane>=16 -> kk offset +16: 16 contiguous halves (LDS).
__device__ __forceinline__ v16h ld_bt16(const _Float16* base, int ncol0,
                                        int ldh, int k0) {
  const int lane = threadIdx.x & 31;
  const _Float16* p = base + (ncol0 + (lane & 15)) * ldh + k0 + ((lane >> 4) << 4);
  union { v16h v; uint4 q[2]; } u;
  u.q[0] = *(const uint4*)(p);
  u.q[1] = *(const uint4*)(p + 8);
  return u.v;
}

// B fragment from a pre-converted global f16 weight matrix (row-major
// [Nout, K]): B[kk][n] = W[(n0+n)*ldw + k0 + kk].  Two global b128 loads.
__device__ __forceinline__ v16h ld_bw_h(const _Float16* __restrict__ Wt,
                                        int n0, int ldw, int k0) {
  const int lane = threadIdx.x & 31;
  const _Float16* p =
      Wt + (size_t)(n0 + (lane & 15)) * ldw + k0 + ((lane >> 4) << 4);
  union { v16h v; uint4 q[2]; } u;
  u.q[0] = *(const uint4*)(p);
  u.q[1] = *(const uint4*)(p + 8);
  return u.v;
}

// A fragment from f32 attention-prob matrix S[98][112]; rows clamped (rows
// >=98 are discarded downstream), K >= 112 returns 0 (zero padding).
__device__ __forceinline__ v16h ld_a_probs(const float* S, int row0, int k0) {
  const int lane = threadIdx.x & 31;
  int r = row0 + (lane & 15);
  if (r > NTOK - 1) r = NTOK - 1;
  const int khi = (lane >> 4) << 4;
  v16h a;
#pragma unroll
  for (int i = 0; i < 16; ++i) {
    const int v = i >> 1, c = i & 1;
    const int K = k0 + 2 * v + c + khi;
    a[i] = (K < NPAD) ? (_Float16)S[r * NPAD + K] : (_Float16)0.0f;
  }
  return a;
}

// ---------------------------------------------------------------------------
// Kernel 0: one-shot f32 -> f16 conversion of qkv / proj weights into d_ws.
// ---------------------------------------------------------------------------
__global__ void wcvt_kernel(const float* __restrict__ qkvw,
                            const float* __restrict__ projw,
                            _Float16* __restrict__ qkvw_h,
                            _Float16* __restrict__ projw_h) {
  const int i = blockIdx.x * 256 + threadIdx.x;
  if (i < 3 * CC * CC) qkvw_h[i] = (_Float16)qkvw[i];
  if (i < CC * CC) projw_h[i] = (_Float16)projw[i];
}

// ---------------------------------------------------------------------------
// Kernel 1: CPB MLP  ->  bias_tab[m][h]  (m in [0,507), h in [0,4))
// ---------------------------------------------------------------------------
__device__ __forceinline__ float cpb_coord(float raw) {
  const float v = 8.0f * raw;
  const float s = (v > 0.0f) ? 1.0f : ((v < 0.0f) ? -1.0f : 0.0f);
  return s * __log2f(fabsf(v) + 1.0f) * (1.0f / 3.0f);
}

__global__ void cpb_mlp_kernel(const float* __restrict__ w1,
                               const float* __restrict__ b1,
                               const float* __restrict__ w2,
                               float* __restrict__ bias_tab) {
  const int m = blockIdx.x;
  const int a = m / 169, b = (m / 13) % 13, c = m % 13;
  const float t0 = cpb_coord((float)(a - 1) / (float)(WD - 1));
  const float t1 = cpb_coord((float)(b - 6) / (float)(WH - 1));
  const float t2 = cpb_coord((float)(c - 6) / (float)(WWIN - 1));

  __shared__ float red[128 * NH];
  float acc[NH] = {0.f, 0.f, 0.f, 0.f};
  for (int u = threadIdx.x; u < 512; u += 128) {
    float hv = t0 * w1[u * 3] + t1 * w1[u * 3 + 1] + t2 * w1[u * 3 + 2] + b1[u];
    hv = fmaxf(hv, 0.0f);
#pragma unroll
    for (int h = 0; h < NH; ++h) acc[h] += hv * w2[h * 512 + u];
  }
#pragma unroll
  for (int h = 0; h < NH; ++h) red[threadIdx.x * NH + h] = acc[h];
  __syncthreads();
  if (threadIdx.x < NH) {
    float s = 0.f;
    for (int i = 0; i < 128; ++i) s += red[i * NH + threadIdx.x];
    bias_tab[m * NH + threadIdx.x] = s;
  }
}

// ---------------------------------------------------------------------------
// Kernel 2: gather + 16*sigmoid  ->  rpb[h][i][j]   (4 x 98 x 98)
// ---------------------------------------------------------------------------
__global__ void rpb_kernel(const float* __restrict__ bias_tab,
                           float* __restrict__ rpb) {
  const int g = blockIdx.x * 256 + threadIdx.x;
  if (g >= NH * NTOK * NTOK) return;
  const int h = g / (NTOK * NTOK);
  const int ij = g % (NTOK * NTOK);
  const int i = ij / NTOK, j = ij % NTOK;
  const int di = i / 49 - j / 49 + (WD - 1);
  const int dh = (i / 7) % 7 - (j / 7) % 7 + (WH - 1);
  const int dw = i % 7 - j % 7 + (WWIN - 1);
  const int idx = di * 169 + dh * 13 + dw;
  const float v = bias_tab[idx * NH + h];
  rpb[g] = 16.0f / (1.0f + __expf(-v));
}

// ---------------------------------------------------------------------------
// Kernel 3: fused shifted-window attention (one block per window)
// ---------------------------------------------------------------------------
__launch_bounds__(256, 1)
__global__ void swin_attn_kernel(const float* __restrict__ x,
                                 const _Float16* __restrict__ qkvw_h,
                                 const float* __restrict__ qkvb,
                                 const _Float16* __restrict__ projw_h,
                                 const float* __restrict__ projb,
                                 const float* __restrict__ lsc,
                                 const float* __restrict__ rpb,
                                 float* __restrict__ out) {
  __shared__ __align__(16) _Float16 sQ[NPAD * CC];    // 28672 B
  __shared__ __align__(16) _Float16 sK[NPAD * CC];    // 28672 B
  __shared__ __align__(16) _Float16 sVT[CC * NPAD];   // 28672 B (transposed V)
  __shared__ __align__(16) float    sS[NTOK * NPAD];  // 43904 B (scores/probs)
  __shared__ __align__(16) _Float16 sXO[NPAD * CC];   // 28672 B (x, later out)
  __shared__ int sREG[NPAD];

  const int lane = threadIdx.x & 31;
  const int wv   = threadIdx.x >> 5;

  const int bw = blockIdx.x;
  const int iw = bw & 15;
  const int ih = (bw >> 4) & 15;
  const int idd = bw >> 8;            // 0..7

  // L2 warm-up hint for the f16 weights this block will stream repeatedly.
  __builtin_prefetch(qkvw_h + (size_t)threadIdx.x * 384, 0, 1);
  __builtin_prefetch(projw_h + (size_t)threadIdx.x * 64, 0, 1);

  // ---- Stage A: gather shifted window -> sXO (f16), region ids -----------
  for (int idx = threadIdx.x; idx < NPAD * 32; idx += 256) {
    const int row = idx >> 5;
    const int cq = (idx & 31) << 2;
    _Float16* dst = &sXO[row * CC + cq];
    if (row < NTOK) {
      const int ld = row / 49, lh = (row / 7) % 7, lw = row % 7;
      const int gd = (idd * WD + ld + SD) & (DD - 1);
      const int gh = (ih * WH + lh + SH) % HH;
      const int gw = (iw * WWIN + lw + SW) % WW_;
      const float4 xv =
          *(const float4*)(x + (((size_t)gd * HH + gh) * WW_ + gw) * CC + cq);
      dst[0] = (_Float16)xv.x; dst[1] = (_Float16)xv.y;
      dst[2] = (_Float16)xv.z; dst[3] = (_Float16)xv.w;
      if (cq == 0) {
        const int rd = (idd == 7)  ? ((ld < WD - SD) ? 1 : 2) : 0;
        const int rh = (ih == 15)  ? ((lh < WH - SH) ? 1 : 2) : 0;
        const int rw = (iw == 15)  ? ((lw < WWIN - SW) ? 1 : 2) : 0;
        sREG[row] = rd * 16 + rh * 4 + rw;
      }
    } else {
      dst[0] = dst[1] = dst[2] = dst[3] = (_Float16)0.0f;
      if (cq == 0) sREG[row] = 0;
    }
  }
  __syncthreads();

  // ---- Stage B: QKV = X @ Wqkv^T + b  (168 tiles of 16x16, K=128) --------
  for (int t = wv; t < 7 * 24; t += 8) {
    const int mt = t % 7, nt = t / 7;
    v8f acc = vzero8();
#pragma unroll
    for (int kt = 0; kt < 4; ++kt) {
      const v16h a = ld_a16(sXO, 16 * mt, CC, 32 * kt);
      const v16h b = ld_bw_h(qkvw_h, 16 * nt, CC, 32 * kt);
      acc = wmma16(a, b, acc);
    }
    const int n = 16 * nt + (lane & 15);
    const float bval = (n >= CC && n < 2 * CC) ? 0.0f : qkvb[n];
    const int mrow = 16 * mt + ((lane >> 4) << 3);
    if (nt < 16) {
      _Float16* dstb = (nt < 8) ? (sQ + n) : (sK + (n - CC));
#pragma unroll
      for (int r2 = 0; r2 < 8; ++r2)
        dstb[(mrow + r2) * CC] = (_Float16)(acc[r2] + bval);
    } else {
      const int f = n - 2 * CC;                       // V feature
#pragma unroll
      for (int r2 = 0; r2 < 8; ++r2)
        sVT[f * NPAD + mrow + r2] = (_Float16)(acc[r2] + bval);
    }
  }
  __syncthreads();

  // ---- Stage C: cosine-normalize q,k rows per head -----------------------
  for (int task = threadIdx.x; task < NTOK * NH; task += 256) {
    const int tk = task >> 2, h = task & 3;
    _Float16* q = sQ + tk * CC + h * HDIM;
    _Float16* k = sK + tk * CC + h * HDIM;
    float sq = 0.f, sk = 0.f;
#pragma unroll
    for (int i = 0; i < HDIM; ++i) {
      const float a = (float)q[i], b = (float)k[i];
      sq += a * a; sk += b * b;
    }
    const float iq = 1.0f / fmaxf(sqrtf(sq), 1e-12f);
    const float ik = 1.0f / fmaxf(sqrtf(sk), 1e-12f);
#pragma unroll
    for (int i = 0; i < HDIM; ++i) {
      q[i] = (_Float16)((float)q[i] * iq);
      k[i] = (_Float16)((float)k[i] * ik);
    }
  }
  __syncthreads();

  // ---- Stage D: per-head attention --------------------------------------
  for (int h = 0; h < NH; ++h) {
    // D1: scores S = qn @ kn^T  (49 tiles, single K=32 WMMA each)
    for (int t = wv; t < 49; t += 8) {
      const int mt = t / 7, nt = t % 7;
      const v16h a = ld_a16(sQ + h * HDIM, 16 * mt, CC, 0);
      const v16h b = ld_bt16(sK + h * HDIM, 16 * nt, CC, 0);
      v8f acc = vzero8();
      acc = wmma16(a, b, acc);
      const int col = 16 * nt + (lane & 15);
      const int mrow = 16 * mt + ((lane >> 4) << 3);
#pragma unroll
      for (int r2 = 0; r2 < 8; ++r2) {
        const int M = mrow + r2;
        if (M < NTOK) sS[M * NPAD + col] = acc[r2];
      }
    }
    __syncthreads();

    // D2: softmax (scale * S + rpb + mask), probs back into sS (f32).
    const float lscale = __expf(fminf(lsc[h], 4.6051702f));  // ln(100)
    for (int t = wv; t < NTOK; t += 8) {
      const int rt = sREG[t];
      float vals[4];
      float m = -1e30f;
#pragma unroll
      for (int it = 0; it < 4; ++it) {
        const int j = lane + (it << 5);
        float v;
        if (j < NTOK) {
          v = sS[t * NPAD + j] * lscale + rpb[(h * NTOK + t) * NTOK + j];
          if (sREG[j] != rt) v -= 100.0f;
        } else {
          v = -1e30f;
        }
        vals[it] = v;
        m = fmaxf(m, v);
      }
#pragma unroll
      for (int off = 16; off; off >>= 1) m = fmaxf(m, __shfl_xor(m, off, 32));
      float sum = 0.f;
#pragma unroll
      for (int it = 0; it < 4; ++it) {
        const float e = (vals[it] > -1e29f) ? __expf(vals[it] - m) : 0.0f;
        vals[it] = e;
        sum += e;
      }
#pragma unroll
      for (int off = 16; off; off >>= 1) sum += __shfl_xor(sum, off, 32);
      const float inv = 1.0f / sum;
#pragma unroll
      for (int it = 0; it < 4; ++it) {
        const int j = lane + (it << 5);
        if (j < NPAD) sS[t * NPAD + j] = vals[it] * inv;
      }
    }
    __syncthreads();

    // D3: out_h = P @ v_h  (14 tiles, 4 K-steps; key padding handled)
    for (int t = wv; t < 14; t += 8) {
      const int mt = t % 7, ntl = t / 7;
      v8f acc = vzero8();
#pragma unroll
      for (int kt = 0; kt < 4; ++kt) {
        const v16h a = ld_a_probs(sS, 16 * mt, 32 * kt);
        v16h b;
        if (kt == 3 && (lane >> 4))
          b = vzero16();                 // keys 112..127 are zero padding
        else
          b = ld_bt16(sVT, h * HDIM + 16 * ntl, NPAD, 32 * kt);
        acc = wmma16(a, b, acc);
      }
      const int ncol = h * HDIM + 16 * ntl + (lane & 15);
      const int mrow = 16 * mt + ((lane >> 4) << 3);
#pragma unroll
      for (int r2 = 0; r2 < 8; ++r2)
        sXO[(mrow + r2) * CC + ncol] = (_Float16)acc[r2];
    }
    __syncthreads();
  }

  // ---- Stage E: Y = O @ Wproj^T + b, scatter with reverse shift ----------
  for (int t = wv; t < 7 * 8; t += 8) {
    const int mt = t % 7, nt = t / 7;
    v8f acc = vzero8();
#pragma unroll
    for (int kt = 0; kt < 4; ++kt) {
      const v16h a = ld_a16(sXO, 16 * mt, CC, 32 * kt);
      const v16h b = ld_bw_h(projw_h, 16 * nt, CC, 32 * kt);
      acc = wmma16(a, b, acc);
    }
    const int n = 16 * nt + (lane & 15);
    const float pb = projb[n];
    const int mbase = 16 * mt + ((lane >> 4) << 3);
#pragma unroll
    for (int r2 = 0; r2 < 8; ++r2) {
      const int M = mbase + r2;
      if (M < NTOK) {
        const int ld = M / 49, lh = (M / 7) % 7, lw = M % 7;
        const int gd = (idd * WD + ld + SD) & (DD - 1);
        const int gh = (ih * WH + lh + SH) % HH;
        const int gw = (iw * WWIN + lw + SW) % WW_;
        out[(((size_t)gd * HH + gh) * WW_ + gw) * CC + n] = acc[r2] + pb;
      }
    }
  }
}

// ---------------------------------------------------------------------------
extern "C" void kernel_launch(void* const* d_in, const int* in_sizes, int n_in,
                              void* d_out, int out_size, void* d_ws,
                              size_t ws_size, hipStream_t stream) {
  const float* x     = (const float*)d_in[0];
  const float* qkvw  = (const float*)d_in[1];
  const float* qkvb  = (const float*)d_in[2];
  const float* projw = (const float*)d_in[3];
  const float* projb = (const float*)d_in[4];
  const float* lsc   = (const float*)d_in[5];
  const float* w1    = (const float*)d_in[6];
  const float* b1    = (const float*)d_in[7];
  const float* w2    = (const float*)d_in[8];
  float* out = (float*)d_out;

  // Workspace layout (bytes):
  //   [0,      8192)   bias_tab  (507*4 floats, padded)
  //   [8192, 161856)   rpb       (4*98*98 floats)    -> ends 161856
  //   [163840,262144)  qkvw_h    (384*128 f16 = 98304 B)
  //   [262144,294912)  projw_h   (128*128 f16 = 32768 B)
  float* bias_tab    = (float*)d_ws;
  float* rpb         = bias_tab + 2048;
  _Float16* qkvw_h   = (_Float16*)((char*)d_ws + 163840);
  _Float16* projw_h  = (_Float16*)((char*)d_ws + 262144);

  wcvt_kernel<<<(3 * CC * CC + 255) / 256, 256, 0, stream>>>(qkvw, projw,
                                                             qkvw_h, projw_h);
  cpb_mlp_kernel<<<MTAB, 128, 0, stream>>>(w1, b1, w2, bias_tab);
  rpb_kernel<<<(NH * NTOK * NTOK + 255) / 256, 256, 0, stream>>>(bias_tab, rpb);
  swin_attn_kernel<<<NWIN, 256, 0, stream>>>(x, qkvw_h, qkvb, projw_h, projb,
                                             lsc, rpb, out);
}
